// MambaSSM_28467043238499
// MI455X (gfx1250) — matrix operand order repaired and checked by reading, not compiled, not run on hardware
//
#include <hip/hip_runtime.h>
#include <hip/hip_bf16.h>

// ---------------------------------------------------------------------------
// MambaSSM for MI455X (gfx1250): bf16 WMMA GEMMs (fp32 accum) + fused SSM
// parameterization epilogue + chunked parallel scan + skinny out-projection.
// GEMM uses 2x2 register blocking (32x32 tile x 4 weight matrices per wave)
// to raise WMMA:VMEM ratio and halve L2 fragment traffic.
// ---------------------------------------------------------------------------

#define T_DIM   8192
#define OBS     512
#define HID     2048
#define NOBJ    8
#define NCH     64            // scan chunks
#define CHL     (T_DIM/NCH)   // 128 steps per chunk

typedef __attribute__((ext_vector_type(16))) __bf16       v16bf;
typedef __attribute__((ext_vector_type(8)))  float        v8f;
typedef __attribute__((ext_vector_type(4)))  unsigned int v4u;

union Frag { v16bf v; v4u u[2]; };

// round-to-nearest-even f32 -> bf16
__device__ __forceinline__ unsigned short f32_to_bf16_rne(float f) {
    unsigned int u = __float_as_uint(f);
    unsigned int r = u + 0x7FFFu + ((u >> 16) & 1u);
    if ((u & 0x7FFFFFFFu) > 0x7F800000u) r = u | 0x00400000u;  // quiet NaN
    return (unsigned short)(r >> 16);
}

__global__ __launch_bounds__(256) void cast_bf16_kernel(const float* __restrict__ src,
                                                        unsigned short* __restrict__ dst,
                                                        int n) {
    int i = blockIdx.x * 256 + threadIdx.x;
    if (i < n) dst[i] = f32_to_bf16_rne(src[i]);
}

// ---------------------------------------------------------------------------
// Fused 4-projection GEMM, 2x2 register-blocked.
// One wave owns a 32x32 (t,h) tile for all four weight matrices:
//   16 fp32 accumulators, 2 A-frags + 8 B-frags per K-step, 16 WMMAs/K-step.
// Epilogue fuses delta = sigmoid(d), A_bar = exp(delta * -exp(A_log[h])),
// Bx = B * x_state, so only A_bar / Bx / C reach HBM.
//
// Fragment layouts per CDNA5 ISA 7.12.2 (wave32):
//   A (16x32 bf16): lane L -> row M = L%16; elems 0..7  : K = kk + 8*(L/16)+i
//                                          elems 8..15 : K = kk+16+8*(L/16)+i
//   B (32x16 bf16): lane L -> col N = L%16; same K order as A
//   C/D (16x16 f32): vgpr r, lane L -> M = r + 8*(L/16), N = L%16
// ---------------------------------------------------------------------------
__global__ __launch_bounds__(256) void gemm4_wmma_kernel(
        const unsigned short* __restrict__ xb,      // [T, OBS] bf16
        const unsigned short* __restrict__ wb,      // 4 x [HID, OBS] bf16
        const float* __restrict__ A_log,            // [HID]
        float* __restrict__ Abar,                   // [T, HID]
        float* __restrict__ Bx,                     // [T, HID]
        float* __restrict__ Cp)                     // [T, HID]
{
    const int lane   = threadIdx.x & 31;
    const int wave   = threadIdx.x >> 5;
    const int gid    = blockIdx.x * 8 + wave;      // 16384 waves total
    const int hb     = gid & 63;                   // HID/32 blocks
    const int mb     = gid >> 6;                   // T/32 blocks
    const int laneLo = lane & 15;
    const int laneHi = lane >> 4;

    const unsigned short* aRow[2];
    aRow[0] = xb + (size_t)(mb * 32 + laneLo) * OBS;
    aRow[1] = aRow[0] + (size_t)16 * OBS;
    const unsigned short* bRow[2];
    bRow[0] = wb + (size_t)(hb * 32 + laneLo) * OBS;
    bRow[1] = bRow[0] + (size_t)16 * OBS;

    v8f acc[2][2][4];
    #pragma unroll
    for (int mi = 0; mi < 2; ++mi)
        #pragma unroll
        for (int ni = 0; ni < 2; ++ni)
            #pragma unroll
            for (int m = 0; m < 4; ++m)
                acc[mi][ni][m] = (v8f){};

    for (int kk = 0; kk < OBS; kk += 32) {
        Frag af[2];
        #pragma unroll
        for (int mi = 0; mi < 2; ++mi) {
            const unsigned short* ap = aRow[mi] + kk + 8 * laneHi;
            af[mi].u[0] = *(const v4u*)(ap);
            af[mi].u[1] = *(const v4u*)(ap + 16);
        }

        Frag bfv[2][4];
        #pragma unroll
        for (int ni = 0; ni < 2; ++ni) {
            const unsigned short* bp = bRow[ni] + kk + 8 * laneHi;
            #pragma unroll
            for (int m = 0; m < 4; ++m) {
                bfv[ni][m].u[0] = *(const v4u*)(bp + (size_t)m * HID * OBS);
                bfv[ni][m].u[1] = *(const v4u*)(bp + (size_t)m * HID * OBS + 16);
            }
        }

        #pragma unroll
        for (int mi = 0; mi < 2; ++mi)
            #pragma unroll
            for (int ni = 0; ni < 2; ++ni)
                #pragma unroll
                for (int m = 0; m < 4; ++m)
                    acc[mi][ni][m] = __builtin_amdgcn_wmma_f32_16x16x32_bf16(
                        false, af[mi].v, false, bfv[ni][m].v,
                        (short)0, acc[mi][ni][m], false, false);
    }

    // fused SSM-parameter epilogue
    #pragma unroll
    for (int ni = 0; ni < 2; ++ni) {
        const int   h    = hb * 32 + ni * 16 + laneLo;
        const float aneg = -__expf(A_log[h]);
        #pragma unroll
        for (int mi = 0; mi < 2; ++mi) {
            #pragma unroll
            for (int r = 0; r < 8; ++r) {
                const int    trow = mb * 32 + mi * 16 + r + 8 * laneHi;
                const size_t idx  = (size_t)trow * HID + h;
                const float  xs = acc[mi][ni][0][r];
                const float  bv = acc[mi][ni][1][r];
                const float  cv = acc[mi][ni][2][r];
                const float  dv = acc[mi][ni][3][r];
                const float  delta = 1.0f / (1.0f + __expf(-dv));
                Abar[idx] = __expf(delta * aneg);
                Bx[idx]   = bv * xs;
                Cp[idx]   = cv;
            }
        }
    }
}

// ---------------------------------------------------------------------------
// Chunked parallel scan:  h_t = a_t * h_{t-1} + bx_t
// pass 1: per-(chunk, channel) local scan -> (prod of a, local end state)
// pass 2: tiny serial carry over 64 chunk summaries per channel
// pass 3: replay with corrected initial state, fuse y = C * h
// ---------------------------------------------------------------------------
__global__ __launch_bounds__(256) void scan_pass1_kernel(
        const float* __restrict__ Abar, const float* __restrict__ Bx,
        float* __restrict__ P, float* __restrict__ S)
{
    const int tid = blockIdx.x * 256 + threadIdx.x;   // NCH*HID threads
    const int h   = tid & (HID - 1);
    const int c   = tid >> 11;

    const float* ap = Abar + (size_t)c * CHL * HID + h;
    const float* bp = Bx   + (size_t)c * CHL * HID + h;

    float p = 1.0f, s = 0.0f;
    for (int t = 0; t < CHL; ++t) {
        const float a = ap[(size_t)t * HID];
        const float b = bp[(size_t)t * HID];
        s = fmaf(a, s, b);
        p *= a;
    }
    P[tid] = p;
    S[tid] = s;
}

__global__ __launch_bounds__(256) void scan_pass2_kernel(
        const float* __restrict__ P, const float* __restrict__ S,
        float* __restrict__ Hinit)
{
    const int h = blockIdx.x * 256 + threadIdx.x;     // HID threads
    float carry = 0.0f;
    for (int c = 0; c < NCH; ++c) {
        const size_t i = (size_t)c * HID + h;
        Hinit[i] = carry;
        carry = fmaf(P[i], carry, S[i]);
    }
}

__global__ __launch_bounds__(256) void scan_pass3_kernel(
        const float* __restrict__ Abar, const float* __restrict__ Bx,
        const float* __restrict__ Cp, const float* __restrict__ Hinit,
        float* __restrict__ Y)
{
    const int tid = blockIdx.x * 256 + threadIdx.x;   // NCH*HID threads
    const int h   = tid & (HID - 1);
    const int c   = tid >> 11;

    const size_t base = (size_t)c * CHL * HID + h;
    float s = Hinit[tid];
    for (int t = 0; t < CHL; ++t) {
        const size_t i = base + (size_t)t * HID;
        s = fmaf(Abar[i], s, Bx[i]);
        Y[i] = Cp[i] * s;
    }
}

// ---------------------------------------------------------------------------
// out[t,j] = sum_h Y[t,h]*W_out[j,h] + b_out[j] + sum_o x[t,o]*W_skip[j,o]
// One block per time step, wave-shuffle reduction (wave32).
// ---------------------------------------------------------------------------
__global__ __launch_bounds__(128) void outproj_kernel(
        const float* __restrict__ Y, const float* __restrict__ x,
        const float* __restrict__ W_out, const float* __restrict__ b_out,
        const float* __restrict__ W_skip, float* __restrict__ out)
{
    const int t   = blockIdx.x;
    const int tid = threadIdx.x;

    float acc[NOBJ];
    #pragma unroll
    for (int j = 0; j < NOBJ; ++j) acc[j] = 0.0f;

    const float* yr = Y + (size_t)t * HID;
    for (int h = tid; h < HID; h += 128) {
        const float yv = yr[h];
        #pragma unroll
        for (int j = 0; j < NOBJ; ++j)
            acc[j] = fmaf(yv, W_out[j * HID + h], acc[j]);
    }
    const float* xr = x + (size_t)t * OBS;
    for (int o = tid; o < OBS; o += 128) {
        const float xv = xr[o];
        #pragma unroll
        for (int j = 0; j < NOBJ; ++j)
            acc[j] = fmaf(xv, W_skip[j * OBS + o], acc[j]);
    }

    // intra-wave reduction (wave32)
    #pragma unroll
    for (int j = 0; j < NOBJ; ++j)
        #pragma unroll
        for (int off = 16; off > 0; off >>= 1)
            acc[j] += __shfl_xor(acc[j], off, 32);

    __shared__ float sm[4][NOBJ];
    const int w = tid >> 5, l = tid & 31;
    if (l == 0) {
        #pragma unroll
        for (int j = 0; j < NOBJ; ++j) sm[w][j] = acc[j];
    }
    __syncthreads();
    if (tid < NOBJ)
        out[(size_t)t * NOBJ + tid] =
            sm[0][tid] + sm[1][tid] + sm[2][tid] + sm[3][tid] + b_out[tid];
}

// ---------------------------------------------------------------------------

extern "C" void kernel_launch(void* const* d_in, const int* in_sizes, int n_in,
                              void* d_out, int out_size, void* d_ws, size_t ws_size,
                              hipStream_t stream) {
    (void)in_sizes; (void)n_in; (void)out_size; (void)ws_size;

    const float* x       = (const float*)d_in[0];
    const float* W_in    = (const float*)d_in[1];
    const float* W_B     = (const float*)d_in[2];
    const float* W_C     = (const float*)d_in[3];
    const float* W_delta = (const float*)d_in[4];
    const float* A_log   = (const float*)d_in[5];
    const float* W_out   = (const float*)d_in[6];
    const float* b_out   = (const float*)d_in[7];
    const float* W_skip  = (const float*)d_in[8];
    float*       out     = (float*)d_out;

    // workspace layout (bytes)
    char* ws = (char*)d_ws;
    const size_t thSz = (size_t)T_DIM * HID * sizeof(float);   // 64 MB
    const size_t oAbar = 0;
    const size_t oBx   = oAbar + thSz;
    const size_t oC    = oBx + thSz;
    const size_t oY    = oC + thSz;
    const size_t oXb   = oY + thSz;
    const size_t oWb   = oXb + (size_t)T_DIM * OBS * 2;
    const size_t oP    = oWb + (size_t)4 * HID * OBS * 2;
    const size_t oS    = oP + (size_t)NCH * HID * sizeof(float);
    const size_t oH    = oS + (size_t)NCH * HID * sizeof(float);

    float*          Abar  = (float*)(ws + oAbar);
    float*          Bx    = (float*)(ws + oBx);
    float*          Cp    = (float*)(ws + oC);
    float*          Y     = (float*)(ws + oY);
    unsigned short* xb    = (unsigned short*)(ws + oXb);
    unsigned short* wb    = (unsigned short*)(ws + oWb);
    float*          P     = (float*)(ws + oP);
    float*          S     = (float*)(ws + oS);
    float*          Hinit = (float*)(ws + oH);

    // 1) cast inputs to bf16
    const int nX = T_DIM * OBS;
    const int nW = HID * OBS;
    cast_bf16_kernel<<<(nX + 255) / 256, 256, 0, stream>>>(x, xb, nX);
    cast_bf16_kernel<<<(nW + 255) / 256, 256, 0, stream>>>(W_in,    wb + (size_t)0 * nW, nW);
    cast_bf16_kernel<<<(nW + 255) / 256, 256, 0, stream>>>(W_B,     wb + (size_t)1 * nW, nW);
    cast_bf16_kernel<<<(nW + 255) / 256, 256, 0, stream>>>(W_C,     wb + (size_t)2 * nW, nW);
    cast_bf16_kernel<<<(nW + 255) / 256, 256, 0, stream>>>(W_delta, wb + (size_t)3 * nW, nW);

    // 2) fused 4-way WMMA GEMM + SSM-parameter epilogue
    //    (T/32)*(HID/32) = 16384 waves, 8 waves/block
    gemm4_wmma_kernel<<<2048, 256, 0, stream>>>(xb, wb, A_log, Abar, Bx, Cp);

    // 3) chunked parallel scan
    scan_pass1_kernel<<<(NCH * HID) / 256, 256, 0, stream>>>(Abar, Bx, P, S);
    scan_pass2_kernel<<<HID / 256, 256, 0, stream>>>(P, S, Hinit);
    scan_pass3_kernel<<<(NCH * HID) / 256, 256, 0, stream>>>(Abar, Bx, Cp, Hinit, Y);

    // 4) skinny output projection + skip + bias
    outproj_kernel<<<T_DIM, 128, 0, stream>>>(Y, x, W_out, b_out, W_skip, out);
}
